// ElmoLstm_37469294690804
// MI455X (gfx1250) — matrix operand order repaired and checked by reading, not compiled, not run on hardware
//
#include <hip/hip_runtime.h>

typedef __attribute__((ext_vector_type(16))) __bf16 v16bf;
typedef __attribute__((ext_vector_type(8)))  __bf16 v8bf;
typedef __attribute__((ext_vector_type(8)))  float  v8f;

#define Bsz 64
#define Tt  128
#define Dd  512
#define Hh  512
#define Cc  4096
#define G4  16384   // 4*C
#define KCAT 1024   // D + H fused K dimension
#define NL  2

__device__ __forceinline__ float sigmoidf_(float x) { return 1.0f / (1.0f + __expf(-x)); }
__device__ __forceinline__ float clip3_(float x)    { return fminf(fmaxf(x, -3.0f), 3.0f); }

__device__ __forceinline__ v16bf cat8_(v8bf lo, v8bf hi) {
    return __builtin_shufflevector(lo, hi, 0, 1, 2, 3, 4, 5, 6, 7,
                                           8, 9, 10, 11, 12, 13, 14, 15);
}

// ---------------------------------------------------------------------------
// One-time weight conversion: Wcat[l][d][row][k] (bf16), k<512 from Wi, else Wh
// ---------------------------------------------------------------------------
__global__ void convert_wcat(const float* __restrict__ Wi, const float* __restrict__ Wh,
                             __bf16* __restrict__ Wcat) {
    size_t idx = (size_t)blockIdx.x * blockDim.x + threadIdx.x;
    size_t total = (size_t)NL * 2 * G4 * KCAT;
    if (idx >= total) return;
    int    k     = (int)(idx % KCAT);
    size_t row_g = idx / KCAT;                  // (l*2+d)*G4 + row
    float  v     = (k < Dd) ? Wi[row_g * Dd + k] : Wh[row_g * Hh + (k - Dd)];
    Wcat[idx] = (__bf16)v;
}

__global__ void convert_wp(const float* __restrict__ Wp, __bf16* __restrict__ Wpb) {
    size_t idx = (size_t)blockIdx.x * blockDim.x + threadIdx.x;
    size_t total = (size_t)NL * 2 * Hh * Cc;
    if (idx >= total) return;
    Wpb[idx] = (__bf16)Wp[idx];
}

// Per-layer input -> bf16 staging: Xbf[dir][b][t][512]
__global__ void convert_xin(const float* __restrict__ Xin, int xstride, int xdiroff,
                            __bf16* __restrict__ Xbf) {
    int idx = blockIdx.x * blockDim.x + threadIdx.x;   // 2*B*T*512 = 8388608
    if (idx >= 2 * Bsz * Tt * Dd) return;
    int j   = idx % Dd;
    int t   = (idx / Dd) % Tt;
    int bm  = (idx / (Dd * Tt)) % Bsz;
    int dir = idx / (Dd * Tt * Bsz);
    Xbf[idx] = (__bf16)Xin[(size_t)(bm * Tt + t) * xstride + dir * xdiroff + j];
}

__global__ void zero_f32(float* __restrict__ p, int n) {
    int i = blockIdx.x * blockDim.x + threadIdx.x;
    if (i < n) p[i] = 0.0f;
}

// ---------------------------------------------------------------------------
// Gates GEMM: gates[dir][64][16384] = [x_t | h] @ Wcat^T + bias
// Block = (dir, mt): stages its 16x1024 bf16 A-strip in LDS once, 8 waves each
// compute a 16x64 strip (4 tiles, K-loop 32 WMMAs each).
// Launch: <<<256, 256>>>   bid: ng8 = bid&31, mt = (bid>>5)&3, dir = bid>>7
// ---------------------------------------------------------------------------
__global__ void gates_kernel(const __bf16* __restrict__ Xbf,     // [2][B][T][512]
                             const __bf16* __restrict__ Hbf,     // [2][B][512]
                             const __bf16* __restrict__ Wcat_l,  // [2][G4][KCAT]
                             const float* __restrict__ bias_l,   // [2][G4]
                             float* __restrict__ gates,          // [2][B][G4]
                             int t_fwd)
{
    __shared__ __bf16 smem[16 * KCAT];   // 32 KB A-strip

    int bid = blockIdx.x;
    int ng8 = bid & 31;
    int mt  = (bid >> 5) & 3;
    int dir = bid >> 7;
    int t_eff = dir ? (Tt - 1 - t_fwd) : t_fwd;

    // Cooperative LDS staging of A-strip: rows m = mt*16..+15, k = [x_t | h]
    {
        int tid = threadIdx.x;
        for (int c = tid; c < 16 * KCAT / 8; c += 256) {   // 2048 v8bf chunks
            int row = c >> 7;            // / 128
            int kb  = (c & 127) << 3;    // * 8
            int m   = mt * 16 + row;
            v8bf v;
            if (kb < Dd)
                v = *(const v8bf*)(Xbf + (((size_t)dir * Bsz + m) * Tt + t_eff) * Dd + kb);
            else
                v = *(const v8bf*)(Hbf + ((size_t)dir * Bsz + m) * Hh + (kb - Dd));
            *(v8bf*)(smem + row * KCAT + kb) = v;
        }
    }
    __syncthreads();

    int wv   = threadIdx.x >> 5;
    int lane = threadIdx.x & 31;
    int ml   = lane & 15;
    int hi   = lane >> 4;

    const __bf16* wbase = Wcat_l + (size_t)dir * G4 * KCAT;
    v8f acc[4] = {};

    for (int k0 = 0; k0 < KCAT; k0 += 32) {
        // A fragment: two ds_load_b128 from the strip
        const __bf16* ap = smem + ml * KCAT + k0 + hi * 8;
        v16bf a = cat8_(*(const v8bf*)ap, *(const v8bf*)(ap + 16));
#pragma unroll
        for (int nt = 0; nt < 4; ++nt) {
            int n = (ng8 * 32 + wv * 4 + nt) * 16 + ml;
            const __bf16* wp = wbase + (size_t)n * KCAT + k0 + hi * 8;
            if (k0 + 32 < KCAT) __builtin_prefetch(wp + 32, 0, 1);
            v16bf bf = cat8_(*(const v8bf*)wp, *(const v8bf*)(wp + 16));
            acc[nt] = __builtin_amdgcn_wmma_f32_16x16x32_bf16(
                false, a, false, bf, (short)0, acc[nt], false, false);
        }
    }

    // C/D layout: VGPR r -> row M = hi*8 + r, col N = lane&15
#pragma unroll
    for (int nt = 0; nt < 4; ++nt) {
        int n = (ng8 * 32 + wv * 4 + nt) * 16 + ml;
        float bv = bias_l[dir * G4 + n];
        float* gout = gates + (size_t)dir * Bsz * G4;
#pragma unroll
        for (int r = 0; r < 8; ++r) {
            int mrow = mt * 16 + hi * 8 + r;
            gout[(size_t)mrow * G4 + n] = acc[nt][r] + bv;
        }
    }
}

// ---------------------------------------------------------------------------
// Cell update (elementwise over dir x B x C)
// ---------------------------------------------------------------------------
__global__ void cell_kernel(const float* __restrict__ gates,  // [2][B][G4]
                            float* __restrict__ cbuf,         // [2][B][C]
                            __bf16* __restrict__ abuf)        // [2][B][C]
{
    int idx = blockIdx.x * blockDim.x + threadIdx.x;   // 2*B*C = 524288
    if (idx >= 2 * Bsz * Cc) return;
    int dir = idx / (Bsz * Cc);
    int rem = idx - dir * (Bsz * Cc);
    int bm  = rem / Cc;
    int j   = rem - bm * Cc;
    const float* g = gates + ((size_t)dir * Bsz + bm) * G4;
    float ig = sigmoidf_(g[j]);
    float fg = sigmoidf_(g[Cc + j]);
    float gg = tanhf(g[2 * Cc + j]);
    float og = sigmoidf_(g[3 * Cc + j]);
    float c  = cbuf[idx];
    float cn = clip3_(ig * gg + fg * c);
    cbuf[idx] = cn;
    abuf[idx] = (__bf16)(og * tanhf(cn));
}

// ---------------------------------------------------------------------------
// Projection GEMM: h[dir][64][512] = clip( a[64x4096] @ Wp^T , +-3 )
// Writes fp32 h (state/final), bf16 h (next-step WMMA A operand), seqraw.
// Waves: dir(2) x mt(4) x nt(32) = 256; launch <<<32, 256>>>
// ---------------------------------------------------------------------------
__global__ void proj_kernel(const __bf16* __restrict__ abuf,  // [2][B][C]
                            const __bf16* __restrict__ Wp_l,  // [2][H][C]
                            float* __restrict__ hbuf,         // [2][B][H]
                            __bf16* __restrict__ Hbf,         // [2][B][H]
                            float* __restrict__ seqraw,       // [2][B][T][H]
                            int t_fwd)
{
    int wave = (blockIdx.x * blockDim.x + threadIdx.x) >> 5;
    int lane = threadIdx.x & 31;
    int nt  = wave & 31;
    int mt  = (wave >> 5) & 3;
    int dir = wave >> 7;
    int t_eff = dir ? (Tt - 1 - t_fwd) : t_fwd;

    int ml = lane & 15;
    int hi = lane >> 4;

    int m = mt * 16 + ml;
    int n = nt * 16 + ml;
    const __bf16* arow = abuf + ((size_t)dir * Bsz + m) * Cc;
    const __bf16* wrow = Wp_l + ((size_t)dir * Hh + n) * Cc;

    v8f acc = {};
    for (int k0 = 0; k0 < Cc; k0 += 32) {
        const __bf16* ap = arow + k0 + hi * 8;
        const __bf16* wp = wrow + k0 + hi * 8;
        if (k0 + 32 < Cc) __builtin_prefetch(wp + 32, 0, 1);
        v16bf a  = cat8_(*(const v8bf*)ap, *(const v8bf*)(ap + 16));
        v16bf bf = cat8_(*(const v8bf*)wp, *(const v8bf*)(wp + 16));
        acc = __builtin_amdgcn_wmma_f32_16x16x32_bf16(
            false, a, false, bf, (short)0, acc, false, false);
    }

#pragma unroll
    for (int r = 0; r < 8; ++r) {
        int mrow = mt * 16 + hi * 8 + r;
        float v = clip3_(acc[r]);
        size_t hidx = ((size_t)dir * Bsz + mrow) * Hh + n;
        hbuf[hidx] = v;
        Hbf[hidx]  = (__bf16)v;
        seqraw[(((size_t)dir * Bsz + mrow) * Tt + t_eff) * Hh + n] = v;
    }
}

// ---------------------------------------------------------------------------
// Per-layer epilogue: residual add + write seq_outs slice of d_out
// ---------------------------------------------------------------------------
__global__ void finalize_seq(const float* __restrict__ seqraw,  // [2][B][T][H]
                             const float* __restrict__ Xin, int xstride, int xdiroff,
                             int has_res,
                             float* __restrict__ out_seq)       // [B][T][2H]
{
    int idx = blockIdx.x * blockDim.x + threadIdx.x;  // 2*B*T*H = 8388608
    if (idx >= 2 * Bsz * Tt * Hh) return;
    int j   = idx % Hh;
    int t   = (idx / Hh) % Tt;
    int bm  = (idx / (Hh * Tt)) % Bsz;
    int dir = idx / (Hh * Tt * Bsz);
    float v = seqraw[idx];
    if (has_res) v += Xin[(size_t)(bm * Tt + t) * xstride + dir * xdiroff + j];
    out_seq[(size_t)(bm * Tt + t) * (2 * Hh) + dir * Hh + j] = v;
}

__global__ void copy_hc(const float* __restrict__ hbuf, const float* __restrict__ cbuf,
                        float* __restrict__ out_h,   // [B][2H]
                        float* __restrict__ out_c)   // [B][2C]
{
    int idx = blockIdx.x * blockDim.x + threadIdx.x;
    const int n1 = 2 * Bsz * Hh;
    const int n2 = 2 * Bsz * Cc;
    if (idx < n1) {
        int dir = idx / (Bsz * Hh);
        int rem = idx - dir * (Bsz * Hh);
        int bm = rem / Hh, j = rem - bm * Hh;
        out_h[(size_t)bm * (2 * Hh) + dir * Hh + j] = hbuf[idx];
    } else if (idx < n1 + n2) {
        int k = idx - n1;
        int dir = k / (Bsz * Cc);
        int rem = k - dir * (Bsz * Cc);
        int bm = rem / Cc, j = rem - bm * Cc;
        out_c[(size_t)bm * (2 * Cc) + dir * Cc + j] = cbuf[k];
    }
}

// ---------------------------------------------------------------------------
extern "C" void kernel_launch(void* const* d_in, const int* in_sizes, int n_in,
                              void* d_out, int out_size, void* d_ws, size_t ws_size,
                              hipStream_t stream) {
    const float* x  = (const float*)d_in[0];
    const float* Wi = (const float*)d_in[1];
    const float* Wh = (const float*)d_in[2];
    const float* b  = (const float*)d_in[3];
    const float* Wp = (const float*)d_in[4];
    float* out = (float*)d_out;

    char* ws = (char*)d_ws;
    size_t off = 0;
    __bf16* Wcat = (__bf16*)(ws + off); off += (size_t)NL * 2 * G4 * KCAT * sizeof(__bf16);
    __bf16* Wpb  = (__bf16*)(ws + off); off += (size_t)NL * 2 * Hh * Cc * sizeof(__bf16);
    __bf16* Xbf  = (__bf16*)(ws + off); off += (size_t)2 * Bsz * Tt * Dd * sizeof(__bf16);
    __bf16* Hbf  = (__bf16*)(ws + off); off += (size_t)2 * Bsz * Hh * sizeof(__bf16);
    float* hbuf  = (float*)(ws + off);  off += (size_t)2 * Bsz * Hh * sizeof(float);
    float* cbuf  = (float*)(ws + off);  off += (size_t)2 * Bsz * Cc * sizeof(float);
    float* gates = (float*)(ws + off);  off += (size_t)2 * Bsz * G4 * sizeof(float);
    __bf16* abuf = (__bf16*)(ws + off); off += (size_t)2 * Bsz * Cc * sizeof(__bf16);
    float* seqraw= (float*)(ws + off);  off += (size_t)2 * Bsz * Tt * Hh * sizeof(float);

    const size_t SEQ_OFF = 0;
    const size_t HS_OFF  = (size_t)NL * Bsz * Tt * 2 * Hh;
    const size_t CS_OFF  = HS_OFF + (size_t)NL * Bsz * 2 * Hh;

    // One-time (per launch) bf16 weight staging — resident in 192MB L2 thereafter
    convert_wcat<<<262144, 256, 0, stream>>>(Wi, Wh, Wcat);
    convert_wp<<<32768, 256, 0, stream>>>(Wp, Wpb);

    for (int l = 0; l < NL; ++l) {
        const float* Xin; int xstride, xdiroff;
        if (l == 0) { Xin = x; xstride = Dd; xdiroff = 0; }
        else { Xin = out + SEQ_OFF + (size_t)(l - 1) * Bsz * Tt * 2 * Hh; xstride = 2 * Hh; xdiroff = Hh; }

        convert_xin<<<32768, 256, 0, stream>>>(Xin, xstride, xdiroff, Xbf);
        zero_f32<<<256, 256, 0, stream>>>(hbuf, 2 * Bsz * Hh);
        zero_f32<<<128, 256, 0, stream>>>((float*)Hbf, 2 * Bsz * Hh / 2);
        zero_f32<<<2048, 256, 0, stream>>>(cbuf, 2 * Bsz * Cc);

        const __bf16* Wcat_l = Wcat + (size_t)l * 2 * G4 * KCAT;
        const float*  bias_l = b + (size_t)l * 2 * G4;
        const __bf16* Wp_l   = Wpb + (size_t)l * 2 * Hh * Cc;

        for (int t = 0; t < Tt; ++t) {
            gates_kernel<<<256, 256, 0, stream>>>(Xbf, Hbf, Wcat_l, bias_l, gates, t);
            cell_kernel<<<2048, 256, 0, stream>>>(gates, cbuf, abuf);
            proj_kernel<<<32, 256, 0, stream>>>(abuf, Wp_l, hbuf, Hbf, seqraw, t);
        }

        finalize_seq<<<32768, 256, 0, stream>>>(seqraw, Xin, xstride, xdiroff, (l > 0) ? 1 : 0,
                                                out + SEQ_OFF + (size_t)l * Bsz * Tt * 2 * Hh);
        copy_hc<<<2304, 256, 0, stream>>>(hbuf, cbuf,
                                          out + HS_OFF + (size_t)l * Bsz * 2 * Hh,
                                          out + CS_OFF + (size_t)l * Bsz * 2 * Cc);
    }
    (void)in_sizes; (void)n_in; (void)out_size; (void)ws_size;
}
